// DTWLosslabels_64321430225515
// MI455X (gfx1250) — matrix operand order repaired and checked by reading, not compiled, not run on hardware
//
#include <hip/hip_runtime.h>
#include <hip/hip_bf16.h>

typedef __attribute__((ext_vector_type(16))) _Float16 v16h;
typedef __attribute__((ext_vector_type(8)))  _Float16 v8h;
typedef __attribute__((ext_vector_type(4)))  _Float16 v4h;
typedef __attribute__((ext_vector_type(8)))  float    v8f;
typedef __attribute__((ext_vector_type(4)))  float    v4f;

#define BIGF 1.0e10f
#define Bsz  32
#define Tsz  512
#define Dsz  128

// -------------------------------------------------------------------------
// Kernel 1: row-normalize (cosine) and convert to f16.
// One wave (32 lanes) per row; lane handles 4 contiguous floats.
// grid = (2048, 3), block = 256 (8 waves -> 8 rows per block).
// blockIdx.y selects tensor: 0=TGT, 1=OTH, 2=X.
// -------------------------------------------------------------------------
__global__ void rownorm_f16_kernel(const float* __restrict__ TGT,
                                   const float* __restrict__ OTH,
                                   const float* __restrict__ X,
                                   _Float16* __restrict__ xn_all) {
  const float* src = (blockIdx.y == 0) ? TGT : (blockIdx.y == 1 ? OTH : X);
  _Float16* dst = xn_all + (size_t)blockIdx.y * (Bsz * Tsz * Dsz);

  const int wave = threadIdx.x >> 5;
  const int lane = threadIdx.x & 31;
  const int row  = blockIdx.x * 8 + wave;          // 0 .. B*T-1

  const float* rp = src + (size_t)row * Dsz + lane * 4;
  v4f v = *(const v4f*)rp;
  float s = v.x * v.x + v.y * v.y + v.z * v.z + v.w * v.w;
#pragma unroll
  for (int m = 16; m >= 1; m >>= 1) s += __shfl_xor(s, m, 32);
  const float inv = 1.0f / (sqrtf(s) + 1e-8f);

  v4h h;
  h.x = (_Float16)(v.x * inv);
  h.y = (_Float16)(v.y * inv);
  h.z = (_Float16)(v.z * inv);
  h.w = (_Float16)(v.w * inv);
  *(v4h*)(dst + (size_t)row * Dsz + lane * 4) = h;
}

// -------------------------------------------------------------------------
// WMMA fragment loaders per CDNA5 ISA §7.12.2 (wave32 layouts).
// A (16x32 f16, MxK): lanes 0-15 row M=lane, halves K={0..7,16..23};
//                     lanes 16-31 row M=lane-16, halves K={8..15,24..31}.
// B (32x16 f16, KxN): lanes 0-15 col N=lane, halves K={0..7,16..23};
//                     lanes 16-31 col N=lane-16, halves K={8..15,24..31}.
// Source data is row-major [T][D] f16, D contiguous -> C = A * B^T.
// -------------------------------------------------------------------------
__device__ __forceinline__ v16h load_a_frag(const _Float16* __restrict__ A,
                                            int lane, int m0, int k0) {
  const int half = lane >> 4;
  const int row  = m0 + (lane & 15);
  const int off  = half * 8;
  const _Float16* p = A + (size_t)row * Dsz + k0;
  v8h lo = *(const v8h*)(p + off);       // K = k0+off   .. +7
  v8h hi = *(const v8h*)(p + 16 + off);  // K = k0+16+off .. +7
  v16h r;
#pragma unroll
  for (int t = 0; t < 8; ++t) { r[t] = lo[t]; r[t + 8] = hi[t]; }
  return r;
}

__device__ __forceinline__ v16h load_b_frag(const _Float16* __restrict__ B,
                                            int lane, int n0, int k0) {
  const int half = lane >> 4;
  const int col  = n0 + (lane & 15);
  const _Float16* p = B + (size_t)col * Dsz + k0 + half * 8;
  v8h lo = *(const v8h*)(p);
  v8h hi = *(const v8h*)(p + 16);
  v16h r;
#pragma unroll
  for (int t = 0; t < 8; ++t) { r[t] = lo[t]; r[t + 8] = hi[t]; }
  return r;
}

// -------------------------------------------------------------------------
// Kernel 2: batched cosine-cost via WMMA, stored anti-diagonal-major:
//   Cd[z][(m+n)*512 + m] = 1 - dot(xhat[m], yhat[n])
// grid = (32, 32, 128): x = M-tile, y = N-tile, z = pair*32 + batch.
// block = 32 (one wave, EXEC all-ones for WMMA).
// pairs: 0=(OTH,X) 1=(TGT,X) 2=(OTH,OTH) 3=(TGT,TGT)
// -------------------------------------------------------------------------
__global__ void cost_wmma_kernel(const _Float16* __restrict__ xnT,
                                 const _Float16* __restrict__ xnO,
                                 const _Float16* __restrict__ xnX,
                                 float* __restrict__ Cd_all) {
  const int z = blockIdx.z;
  const int p = z >> 5;
  const int b = z & 31;
  const _Float16* Abase = (p == 0 || p == 2) ? xnO : xnT;
  const _Float16* Bbase = (p <= 1) ? xnX : Abase;
  const _Float16* A = Abase + (size_t)b * Tsz * Dsz;
  const _Float16* Bm = Bbase + (size_t)b * Tsz * Dsz;
  float* __restrict__ Cd = Cd_all + (size_t)z * (1024u * 512u);

  const int m0 = blockIdx.x * 16;
  const int n0 = blockIdx.y * 16;
  const int lane = threadIdx.x;

  v8f acc = {};
#pragma unroll
  for (int k0 = 0; k0 < Dsz; k0 += 32) {
    v16h af = load_a_frag(A, lane, m0, k0);
    v16h bf = load_b_frag(Bm, lane, n0, k0);
    acc = __builtin_amdgcn_wmma_f32_16x16x32_f16(
        /*neg_a=*/false, af, /*neg_b=*/false, bf,
        /*c_mod=*/(short)0, acc, /*reuse_a=*/false, /*reuse_b=*/false);
  }

  // D layout: VGPR r -> M = r + 8*(lane>=16), N = lane%16.
  const int n  = n0 + (lane & 15);
  const int mb = m0 + ((lane >> 4) ? 8 : 0);
#pragma unroll
  for (int r = 0; r < 8; ++r) {
    const int m = mb + r;
    Cd[(size_t)(m + n) * 512 + m] = 1.0f - acc[r];
  }
}

// -------------------------------------------------------------------------
// Kernel 3: soft-DTW wavefront DP. One block per (pair,batch) problem.
// block = 512 threads (16 waves); thread tid owns matrix row i = tid+1.
// Rolling diagonals in LDS; one barrier per anti-diagonal.
// Cost reads are coalesced: diagonal k lives at Cd[(k-2)*512 + (i-1)].
// -------------------------------------------------------------------------
__global__ void softdtw_diag_kernel(const float* __restrict__ Cd_all,
                                    float* __restrict__ raw) {
  __shared__ float buf0[Tsz + 1], buf1[Tsz + 1], buf2[Tsz + 1];
  float* Rp2 = buf0;  // diagonal k-2
  float* Rp  = buf1;  // diagonal k-1
  float* Rn  = buf2;  // diagonal k

  const int z = blockIdx.x;
  const float* __restrict__ Cd = Cd_all + (size_t)z * (1024u * 512u);
  const int tid = threadIdx.x;
  const int i = tid + 1;  // 1..512

  Rp[i]  = BIGF;
  Rp2[i] = BIGF;
  if (tid == 0) { Rp[0] = 0.0f; Rp2[0] = BIGF; }
  __syncthreads();

  for (int k = 1; k <= 1024; ++k) {
    const int j = k - i;
    const bool valid = (j >= 1) && (j <= Tsz);
    const float d = valid ? Cd[(size_t)(k - 2) * 512 + (i - 1)] : 0.0f;
    const float a = Rp[i - 1];   // R[i-1, j]
    const float bb = Rp[i];      // R[i,   j-1]
    const float c = Rp2[i - 1];  // R[i-1, j-1]
    const float m = fminf(a, fminf(bb, c));
    const float sm = m - __logf(__expf(m - a) + __expf(m - bb) + __expf(m - c));
    Rn[i] = valid ? (d + sm) : BIGF;
    if (tid == 0) Rn[0] = BIGF;
    __syncthreads();
    float* t = Rp2; Rp2 = Rp; Rp = Rn; Rn = t;  // rotate (uniform across block)
  }

  if (tid == 0) raw[z] = Rp[Tsz];  // R[512, 512]
}

// -------------------------------------------------------------------------
// Kernel 4: diff = raw_OX - raw_TX - 0.5*raw_OO + 0.5*raw_TT  (XX cancels);
// out = mean((diff - labels)^2). One wave.
// -------------------------------------------------------------------------
__global__ void final_loss_kernel(const float* __restrict__ raw,
                                  const float* __restrict__ labels,
                                  float* __restrict__ out) {
  const int b = threadIdx.x;  // 0..31
  const float diff = raw[b] - raw[32 + b] - 0.5f * raw[64 + b] + 0.5f * raw[96 + b];
  const float e = diff - labels[b];
  float v = e * e;
#pragma unroll
  for (int m = 16; m >= 1; m >>= 1) v += __shfl_xor(v, m, 32);
  if (b == 0) out[0] = v * (1.0f / 32.0f);
}

// -------------------------------------------------------------------------
// Launch. Inputs: d_in[0]=TGT, d_in[1]=OTH, d_in[2]=X (f32, B*T*D each),
// d_in[3]=labels (f32, B). d_out: 1 float.
// Workspace layout:
//   [0, 12MB)        : 3 normalized f16 tensors (TGT, OTH, X)
//   [12MB, 12+256MB) : 4x32 diagonal-major cost matrices (1024*512 f32 each)
//   [+256MB, +512B)  : 128 raw soft-DTW scalars
// -------------------------------------------------------------------------
extern "C" void kernel_launch(void* const* d_in, const int* in_sizes, int n_in,
                              void* d_out, int out_size, void* d_ws, size_t ws_size,
                              hipStream_t stream) {
  const float* TGT    = (const float*)d_in[0];
  const float* OTH    = (const float*)d_in[1];
  const float* X      = (const float*)d_in[2];
  const float* labels = (const float*)d_in[3];
  float* out = (float*)d_out;

  char* ws = (char*)d_ws;
  const size_t xn_elems = (size_t)Bsz * Tsz * Dsz;           // 2,097,152
  _Float16* xn_all = (_Float16*)ws;                          // 3 * 4MB
  _Float16* xnT = xn_all;
  _Float16* xnO = xn_all + xn_elems;
  _Float16* xnX = xn_all + 2 * xn_elems;
  float* Cd  = (float*)(ws + 3 * xn_elems * sizeof(_Float16));     // 256 MB
  float* raw = (float*)(ws + 3 * xn_elems * sizeof(_Float16)
                           + (size_t)128 * 1024 * 512 * sizeof(float));

  // 1) normalize rows -> f16
  rownorm_f16_kernel<<<dim3((Bsz * Tsz) / 8, 3, 1), 256, 0, stream>>>(
      TGT, OTH, X, xn_all);

  // 2) cosine cost matrices via WMMA, diagonal-major
  cost_wmma_kernel<<<dim3(Tsz / 16, Tsz / 16, 128), 32, 0, stream>>>(
      xnT, xnO, xnX, Cd);

  // 3) soft-DTW wavefront DP (128 independent problems)
  softdtw_diag_kernel<<<dim3(128, 1, 1), Tsz, 0, stream>>>(Cd, raw);

  // 4) scalar loss
  final_loss_kernel<<<dim3(1, 1, 1), 32, 0, stream>>>(raw, labels, out);
}
